// UniPhyBackbone_81741817577765
// MI455X (gfx1250) — matrix operand order repaired and checked by reading, not compiled, not run on hardware
//
#include <hip/hip_runtime.h>
#include <hip/hip_bf16.h>
#include <math.h>

#define B_   2
#define C_   128
#define L_   32
#define H_   32
#define W_   32
#define WF   17
#define NTOK 1024        // H*W
#define NH   8
#define HD   16
#define C2   256
#define EPS_ 1e-5f
#define PI2  6.28318530717958647692f

typedef __attribute__((ext_vector_type(16))) _Float16 v16h;
typedef __attribute__((ext_vector_type(8)))  _Float16 v8h;
typedef __attribute__((ext_vector_type(8)))  float    v8f;

union H16 { v16h v; v8h q[2]; _Float16 h[16]; };
union F8  { v8f  v; float f[8]; };

// ---------------------------------------------------------------- wave utils
__device__ __forceinline__ float redmax16(float v) {
#pragma unroll
  for (int m = 1; m < 16; m <<= 1) v = fmaxf(v, __shfl_xor(v, m, 32));
  return v;
}
__device__ __forceinline__ float redsum16(float v) {
#pragma unroll
  for (int m = 1; m < 16; m <<= 1) v += __shfl_xor(v, m, 32);
  return v;
}
__device__ __forceinline__ float redsum32(float v) {
#pragma unroll
  for (int m = 1; m < 32; m <<= 1) v += __shfl_xor(v, m, 32);
  return v;
}

// ------------------------------------------------------------------------
// Stage `halfs` f16 weights from global into LDS using the CDNA5 async
// global->LDS engine (per-lane 16B transfers, tracked with ASYNCcnt).
// Generic LDS pointers carry the LDS byte offset in their low 32 bits.
__device__ __forceinline__ void stage_weights_async(_Float16* lds,
                                                    const _Float16* __restrict__ src,
                                                    int halfs) {
  for (int ch = threadIdx.x; (ch << 3) < halfs; ch += blockDim.x) {
    unsigned loff = (unsigned)(uintptr_t)(lds + (ch << 3));
    const _Float16* g = src + (ch << 3);
    asm volatile("global_load_async_to_lds_b128 %0, %1, off"
                 :: "v"(loff), "v"(g) : "memory");
  }
  asm volatile("s_wait_asynccnt 0x0" ::: "memory");
  __syncthreads();
}

// ------------------------------------------------- generic 16x16 WMMA K-loop
// A: [M,K] row-major f16 in global.  Wt: weight [N,K] row-major f16 in global.
__device__ __forceinline__ v8f wmma_kloop(const _Float16* __restrict__ A, int lda,
                                          const _Float16* __restrict__ Wt, int ldw,
                                          int row0, int col0, int K, v8f acc) {
  const int lane  = threadIdx.x & 31;
  const int m     = row0 + (lane & 15);
  const int n     = col0 + (lane & 15);
  const int base8 = (lane & 16) ? 8 : 0;
  const int baseB = (lane & 16) ? 16 : 0;
  for (int kb = 0; kb < K; kb += 32) {
    H16 a, b;
    a.q[0] = *(const v8h*)(A + (size_t)m * lda + kb + base8);
    a.q[1] = *(const v8h*)(A + (size_t)m * lda + kb + base8 + 16);
    b.v    = *(const v16h*)(Wt + (size_t)n * ldw + kb + baseB);
    acc = __builtin_amdgcn_wmma_f32_16x16x32_f16(false, a.v, false, b.v,
                                                 (short)0, acc, false, false);
  }
  return acc;
}

// Same, but the 16 weight rows for this N-tile are staged in LDS ([16][ldw]).
__device__ __forceinline__ v8f wmma_kloop_ldsB(const _Float16* __restrict__ A, int lda,
                                               const _Float16* ldsW, int ldw,
                                               int row0, int K, v8f acc) {
  const int lane  = threadIdx.x & 31;
  const int m     = row0 + (lane & 15);
  const int nl    = lane & 15;
  const int base8 = (lane & 16) ? 8 : 0;
  const int baseB = (lane & 16) ? 16 : 0;
  for (int kb = 0; kb < K; kb += 32) {
    H16 a, b;
    a.v    = *(const v16h*)(A + (size_t)m * lda + kb + baseB);
    b.v    = *(const v16h*)(ldsW + nl * ldw + kb + baseB);   // ds_load_b128
    acc = __builtin_amdgcn_wmma_f32_16x16x32_f16(false, a.v, false, b.v,
                                                 (short)0, acc, false, false);
  }
  return acc;
}

// ------------------------------------------------------------ weight convert
__global__ void k_cvt(const float* __restrict__ wqkv, const float* __restrict__ wproj,
                      const float* __restrict__ wg, const float* __restrict__ wl,
                      const float* __restrict__ mixw, _Float16* __restrict__ out) {
  int i = blockIdx.x * 256 + threadIdx.x;
  if      (i < 49152)  out[i] = (_Float16)wqkv[i];
  else if (i < 65536)  out[i] = (_Float16)wproj[i - 49152];
  else if (i < 81920)  out[i] = (_Float16)wg[i - 65536];
  else if (i < 98304)  out[i] = (_Float16)wl[i - 81920];
  else if (i < 163840) out[i] = (_Float16)mixw[i - 98304];
}

// -------------------------------------- rfft(W) + complex scan over L (fused)
__global__ void k_scan(const float* __restrict__ x, const float* __restrict__ dt,
                       const float* __restrict__ nu, const float* __restrict__ theta,
                       float2* __restrict__ states) {
  __shared__ float xrow[W_];
  __shared__ float tc[W_], ts[W_];
  const int t = threadIdx.x;                    // 0..31
  const int blk = blockIdx.x;                   // (b*C + c)*H + h
  const int h = blk & 31, c = (blk >> 5) & 127, b = blk >> 12;
  __sincosf(PI2 * (float)t / 32.0f, &ts[t], &tc[t]);
  __syncthreads();
  const int wf = t;
  float nuv = 0.f, thv = 0.f;
  if (wf < WF) { nuv = nu[c * WF + wf]; thv = theta[c * WF + wf]; }
  float sr = 0.f, si = 0.f;
  for (int l = 0; l < L_; ++l) {
    xrow[t] = x[((((size_t)b * C_ + c) * L_ + l) * H_ + h) * W_ + t];
    __syncthreads();
    if (wf < WF) {
      float ur = 0.f, ui = 0.f;
#pragma unroll
      for (int w = 0; w < W_; ++w) {
        int j = (wf * w) & 31;
        float xv = xrow[w];
        ur += xv * tc[j];
        ui -= xv * ts[j];
      }
      float dtv = dt[b * L_ + l];
      float mag = __expf(-nuv * dtv);
      float cs, sn;
      __sincosf(thv * dtv, &sn, &cs);
      float ar = mag * cs, ai = mag * sn;
      float nsr = ar * sr - ai * si + ur;
      float nsi = ar * si + ai * sr + ui;
      sr = nsr; si = nsi;
      states[((((size_t)b * C_ + c) * L_ + l) * H_ + h) * WF + wf] = make_float2(sr, si);
    }
    __syncthreads();
  }
}

// ------------------------------------------- irfft(W) -> token-major xt tensor
__global__ void k_irfft(const float2* __restrict__ states, float* __restrict__ xt) {
  __shared__ float2 srow[8][WF];
  __shared__ float tc[W_], ts[W_];
  const int tid = threadIdx.x;
  if (tid < 32) __sincosf(PI2 * (float)tid / 32.0f, &ts[tid], &tc[tid]);
  const int r = tid >> 5, w = tid & 31;
  const int rg = blockIdx.x * 8 + r;            // (b,c,l,h) flat
  const int h = rg & 31, l = (rg >> 5) & 31, c = (rg >> 10) & 127, b = rg >> 17;
  if (w < WF) srow[r][w] = states[(size_t)rg * WF + w];
  __syncthreads();
  float acc = srow[r][0].x;
#pragma unroll
  for (int k = 1; k < 16; ++k) {
    int j = (k * w) & 31;
    acc += 2.0f * (srow[r][k].x * tc[j] - srow[r][k].y * ts[j]);
  }
  acc += srow[r][16].x * ((w & 1) ? -1.0f : 1.0f);
  acc *= (1.0f / 32.0f);
  const int bn = b * L_ + l, n = h * 32 + w;
  xt[((size_t)bn * NTOK + n) * C_ + c] = acc;
}

// --------------------------------------------------------- LayerNorm -> f16
__global__ void k_ln(const float* __restrict__ xt, const float* __restrict__ g,
                     const float* __restrict__ b, _Float16* __restrict__ xn) {
  const int lane = threadIdx.x & 31;
  const int token = blockIdx.x * 8 + (threadIdx.x >> 5);
  const float4 xv = *(const float4*)(xt + (size_t)token * C_ + lane * 4);
  float s  = xv.x + xv.y + xv.z + xv.w;
  float sq = xv.x * xv.x + xv.y * xv.y + xv.z * xv.z + xv.w * xv.w;
  s = redsum32(s); sq = redsum32(sq);
  float mu = s / 128.0f;
  float rs = rsqrtf(sq / 128.0f - mu * mu + EPS_);
  int c0 = lane * 4;
  _Float16* dst = xn + (size_t)token * C_ + c0;
  dst[0] = (_Float16)((xv.x - mu) * rs * g[c0 + 0] + b[c0 + 0]);
  dst[1] = (_Float16)((xv.y - mu) * rs * g[c0 + 1] + b[c0 + 1]);
  dst[2] = (_Float16)((xv.z - mu) * rs * g[c0 + 2] + b[c0 + 2]);
  dst[3] = (_Float16)((xv.w - mu) * rs * g[c0 + 3] + b[c0 + 3]);
}

// ---------------------- QKV GEMM (WMMA, f16, weights async-staged in LDS)
// block = (mt group of 4, nt); 4 waves share one 16x128 weight tile.
__global__ void k_qkv(const _Float16* __restrict__ xn, const _Float16* __restrict__ wq,
                      _Float16* __restrict__ q, _Float16* __restrict__ k,
                      _Float16* __restrict__ v) {
  __shared__ __align__(16) _Float16 ldsw[16 * 128];
  const int nt = blockIdx.x % 24;
  const int mtg = blockIdx.x / 24;
  stage_weights_async(ldsw, wq + (size_t)nt * 16 * 128, 16 * 128);
  const int wid = threadIdx.x >> 5, lane = threadIdx.x & 31;
  const int mt = mtg * 4 + wid;
  F8 acc;
#pragma unroll
  for (int i = 0; i < 8; ++i) acc.f[i] = 0.f;
  acc.v = wmma_kloop_ldsB(xn, 128, ldsw, 128, mt * 16, 128, acc.v);
  const int off = (lane & 16) ? 8 : 0;
  const int col = nt * 16 + (lane & 15);
  const int e = col >> 7, rem = col & 127, hh = rem >> 4, d = rem & 15;
  _Float16* dst = (e == 0) ? q : (e == 1) ? k : v;
#pragma unroll
  for (int i = 0; i < 8; ++i) {
    int row = mt * 16 + i + off;
    int bn = row >> 10, n = row & 1023;
    dst[(((size_t)bn * NH + hh) * NTOK + n) * HD + d] = (_Float16)acc.f[i];
  }
}

// --------------------------- flash attention: wave = (bn, head, 16-query tile)
__global__ void k_attn(const _Float16* __restrict__ q, const _Float16* __restrict__ kk,
                       const _Float16* __restrict__ v, _Float16* __restrict__ ao) {
  __shared__ __align__(16) _Float16 pbuf[16 * 32];
  const int lane = threadIdx.x;
  const int blk = blockIdx.x;                       // (bn*8+hh)*64 + qt
  const int qt = blk & 63, bh = blk >> 6;
  const int bn = bh >> 3, hh = bh & 7;
  const _Float16* qp = q  + (size_t)bh * NTOK * HD;
  const _Float16* kp = kk + (size_t)bh * NTOK * HD;
  const _Float16* vp = v  + (size_t)bh * NTOK * HD;
  const int base8 = (lane & 16) ? 8 : 0;
  const int row = lane & 15;
  H16 aq;
  if (lane < 16) {
    aq.v = *(const v16h*)(qp + (qt * 16 + row) * HD);  // d = 0..15
  } else {
#pragma unroll
    for (int hx = 0; hx < 16; ++hx) aq.h[hx] = (_Float16)0.0f;   // K 16..31 padded
  }
  F8 acc; float m[8], ssum[8];
#pragma unroll
  for (int i = 0; i < 8; ++i) { acc.f[i] = 0.f; m[i] = -1e30f; ssum[i] = 0.f; }
  for (int c0 = 0; c0 < NTOK; c0 += 32) {
    H16 bk0, bk1;
    if (lane < 16) {
      bk0.v = *(const v16h*)(kp + (c0 + row) * HD);
      bk1.v = *(const v16h*)(kp + (c0 + 16 + row) * HD);
    } else {
#pragma unroll
      for (int hx = 0; hx < 16; ++hx) { bk0.h[hx] = (_Float16)0.0f; bk1.h[hx] = (_Float16)0.0f; }
    }
    F8 s0, s1;
#pragma unroll
    for (int i = 0; i < 8; ++i) { s0.f[i] = 0.f; s1.f[i] = 0.f; }
    s0.v = __builtin_amdgcn_wmma_f32_16x16x32_f16(false, aq.v, false, bk0.v, (short)0, s0.v, false, false);
    s1.v = __builtin_amdgcn_wmma_f32_16x16x32_f16(false, aq.v, false, bk1.v, (short)0, s1.v, false, false);
#pragma unroll
    for (int i = 0; i < 8; ++i) {
      float sa = s0.f[i] * 0.25f, sb = s1.f[i] * 0.25f;  // hd^-0.5
      float t = redmax16(fmaxf(sa, sb));
      float nm = fmaxf(m[i], t);
      float alpha = __expf(m[i] - nm);
      float p0 = __expf(sa - nm), p1 = __expf(sb - nm);
      float rs = redsum16(p0 + p1);
      ssum[i] = ssum[i] * alpha + rs;
      m[i] = nm;
      acc.f[i] *= alpha;
      int r = i + base8;
      pbuf[r * 32 + (lane & 15)]      = (_Float16)p0;
      pbuf[r * 32 + 16 + (lane & 15)] = (_Float16)p1;
    }
    // DS ops are in-order within a wave on CDNA5 -> safe LDS transpose
    H16 ap, bv;
    ap.q[0] = *(const v8h*)(&pbuf[(lane & 15) * 32 + base8]);
    ap.q[1] = *(const v8h*)(&pbuf[(lane & 15) * 32 + base8 + 16]);
    const int baseB = (lane & 16) ? 16 : 0;
#pragma unroll
    for (int hx = 0; hx < 16; ++hx)
      bv.h[hx] = vp[(c0 + base8 + (hx & 7) + ((hx & 8) << 1)) * HD + (lane & 15)];
    acc.v = __builtin_amdgcn_wmma_f32_16x16x32_f16(false, ap.v, false, bv.v, (short)0, acc.v, false, false);
  }
#pragma unroll
  for (int i = 0; i < 8; ++i) {
    int r = qt * 16 + i + base8;
    float ov = acc.f[i] / ssum[i];
    ao[((size_t)bn * NTOK + r) * C_ + hh * 16 + (lane & 15)] = (_Float16)ov;
  }
}

// --------------- proj GEMM + bias + residual (weights async-staged in LDS)
__global__ void k_proj(const _Float16* __restrict__ ao, const _Float16* __restrict__ wp,
                       const float* __restrict__ bp, const float* __restrict__ xt,
                       float* __restrict__ xt2) {
  __shared__ __align__(16) _Float16 ldsw[16 * 128];
  const int nt = blockIdx.x & 7;
  const int mtg = blockIdx.x >> 3;
  stage_weights_async(ldsw, wp + (size_t)nt * 16 * 128, 16 * 128);
  const int wid = threadIdx.x >> 5, lane = threadIdx.x & 31;
  const int mt = mtg * 4 + wid;
  F8 acc;
#pragma unroll
  for (int i = 0; i < 8; ++i) acc.f[i] = 0.f;
  acc.v = wmma_kloop_ldsB(ao, 128, ldsw, 128, mt * 16, 128, acc.v);
  const int off = (lane & 16) ? 8 : 0;
  const int col = nt * 16 + (lane & 15);
#pragma unroll
  for (int i = 0; i < 8; ++i) {
    size_t idx = (size_t)(mt * 16 + i + off) * C_ + col;
    xt2[idx] = xt[idx] + bp[col] + acc.f[i];
  }
}

// ------------------------------------------------------- GroupNorm stats/apply
__global__ void k_gnstats(const float* __restrict__ in, float2* __restrict__ stats) {
  __shared__ float ssum[256], ssq[256];
  const int blk = blockIdx.x;                 // bn*4 + group
  const int bn = blk >> 2, g = blk & 3;
  float s = 0.f, sq = 0.f;
  for (int i = threadIdx.x; i < 32768; i += 256) {
    int n = i >> 5, cc = g * 32 + (i & 31);
    float v = in[((size_t)bn * NTOK + n) * C_ + cc];
    s += v; sq += v * v;
  }
  ssum[threadIdx.x] = s; ssq[threadIdx.x] = sq;
  __syncthreads();
  for (int st = 128; st > 0; st >>= 1) {
    if (threadIdx.x < st) {
      ssum[threadIdx.x] += ssum[threadIdx.x + st];
      ssq[threadIdx.x]  += ssq[threadIdx.x + st];
    }
    __syncthreads();
  }
  if (threadIdx.x == 0) {
    float mu = ssum[0] / 32768.0f;
    float var = ssq[0] / 32768.0f - mu * mu;
    stats[blk] = make_float2(mu, rsqrtf(var + EPS_));
  }
}

__global__ void k_gnapply(const float* __restrict__ in, const float2* __restrict__ stats,
                          const float* __restrict__ g, const float* __restrict__ b,
                          float* __restrict__ out32, _Float16* __restrict__ out16) {
  const int idx = blockIdx.x * 256 + threadIdx.x;   // 8388608
  const int c = idx & 127;
  const int bn = idx >> 17;
  const float2 st = stats[bn * 4 + (c >> 5)];
  float v = (in[idx] - st.x) * st.y * g[c] + b[c];
  out32[idx] = v;
  if (out16) out16[idx] = (_Float16)v;
}

// -------------------------------------------------------------- 2x2 avg pool
__global__ void k_pool(const float* __restrict__ y2n, _Float16* __restrict__ pooled) {
  const int idx = blockIdx.x * 256 + threadIdx.x;   // 64*256*128
  const int c = idx & 127;
  const int pos = (idx >> 7) & 255;
  const int bn = idx >> 15;
  const int hp = pos >> 4, wp = pos & 15;
  const int t0 = (hp * 2) * 32 + wp * 2;
  const size_t base = (size_t)bn * NTOK;
  float v = 0.25f * (y2n[(base + t0) * C_ + c]      + y2n[(base + t0 + 1) * C_ + c] +
                     y2n[(base + t0 + 32) * C_ + c] + y2n[(base + t0 + 33) * C_ + c]);
  pooled[idx] = (_Float16)v;
}

// --------------- g = pooled @ w_g^T (WMMA, weights async-staged in LDS)
__global__ void k_gg(const _Float16* __restrict__ pooled, const _Float16* __restrict__ wg,
                     float* __restrict__ gg) {
  __shared__ __align__(16) _Float16 ldsw[16 * 128];
  const int nt = blockIdx.x & 7;
  const int mtg = blockIdx.x >> 3;
  stage_weights_async(ldsw, wg + (size_t)nt * 16 * 128, 16 * 128);
  const int wid = threadIdx.x >> 5, lane = threadIdx.x & 31;
  const int mt = mtg * 4 + wid;
  F8 acc;
#pragma unroll
  for (int i = 0; i < 8; ++i) acc.f[i] = 0.f;
  acc.v = wmma_kloop_ldsB(pooled, 128, ldsw, 128, mt * 16, 128, acc.v);
  const int off = (lane & 16) ? 8 : 0;
#pragma unroll
  for (int i = 0; i < 8; ++i)
    gg[(size_t)(mt * 16 + i + off) * C_ + nt * 16 + (lane & 15)] = acc.f[i];
}

// ---- l = y2n @ w_l^T (WMMA, full 128x128 w_l async-staged), psi=relu(g+l),
// ---- gate=sigmoid(psi.w_psi + b), y2g = y2n * gate
__global__ void k_gate(const _Float16* __restrict__ y2nh, const _Float16* __restrict__ wl,
                       const float* __restrict__ gg, const float* __restrict__ wpsi,
                       const float* __restrict__ bpsi, const float* __restrict__ y2n,
                       float* __restrict__ y2g) {
  __shared__ __align__(16) _Float16 ldsw[128 * 128];     // 32 KB of the 320KB WGP LDS
  stage_weights_async(ldsw, wl, 128 * 128);
  const int wave = blockIdx.x * 4 + (threadIdx.x >> 5);   // 4096 waves, 16 tokens each
  const int lane = threadIdx.x & 31;
  const int row0 = wave * 16;
  const int off = (lane & 16) ? 8 : 0;
  float part[8];
  int posrow[8];
#pragma unroll
  for (int i = 0; i < 8; ++i) {
    part[i] = 0.f;
    int rowi = row0 + i + off;
    int bn = rowi >> 10, tok = rowi & 1023;
    posrow[i] = (bn * 256 + ((tok >> 5) >> 1) * 16 + ((tok & 31) >> 1)) * 128;
  }
  for (int ct = 0; ct < 8; ++ct) {
    F8 a;
#pragma unroll
    for (int i = 0; i < 8; ++i) a.f[i] = 0.f;
    a.v = wmma_kloop_ldsB(y2nh, 128, ldsw + ct * 16 * 128, 128, row0, 128, a.v);
    const int col = ct * 16 + (lane & 15);
    const float wps = wpsi[col];
#pragma unroll
    for (int i = 0; i < 8; ++i) {
      float psi = fmaxf(a.f[i] + gg[posrow[i] + col], 0.f);
      part[i] += psi * wps;
    }
  }
#pragma unroll
  for (int i = 0; i < 8; ++i) {
    float logit = redsum16(part[i]) + bpsi[0];
    float gate = 1.0f / (1.0f + __expf(-logit));
    const size_t rbase = (size_t)(row0 + i + off) * C_;
#pragma unroll
    for (int ct = 0; ct < 8; ++ct) {
      size_t idx = rbase + ct * 16 + (lane & 15);
      y2g[idx] = y2n[idx] * gate;
    }
  }
}

// ----------------------------------------------- DFT along W (12 low modes)
__global__ void k_dftw(const float* __restrict__ y3, float2* __restrict__ tmp1) {
  const int idx = blockIdx.x * 256 + threadIdx.x;   // ((bn*128+c)*32+h)*12+kw
  const int kw = idx % 12;
  const int rest = idx / 12;
  const int h = rest & 31, c = (rest >> 5) & 127, bn = rest >> 12;
  float sr = 0.f, si = 0.f;
#pragma unroll
  for (int w = 0; w < 32; ++w) {
    float cs, sn;
    __sincosf(PI2 * (float)((kw * w) & 31) / 32.0f, &sn, &cs);
    float xv = y3[((size_t)bn * NTOK + h * 32 + w) * C_ + c];
    sr += xv * cs; si -= xv * sn;
  }
  tmp1[idx] = make_float2(sr, si);
}

// ---------------------- DFT along H, ortho scale, pack z (real||imag) + f16
__global__ void k_dfth(const float2* __restrict__ tmp1, float* __restrict__ z,
                       _Float16* __restrict__ zh) {
  const int idx = blockIdx.x * 256 + threadIdx.x;   // ((bn*128+c)*12+kh)*12+kw
  const int kw = idx % 12;
  const int r2 = idx / 12;
  const int kh = r2 % 12;
  const int r3 = r2 / 12;
  const int c = r3 & 127, bn = r3 >> 7;
  float sr = 0.f, si = 0.f;
#pragma unroll
  for (int h = 0; h < 32; ++h) {
    float cs, sn;
    __sincosf(PI2 * (float)((kh * h) & 31) / 32.0f, &sn, &cs);
    float2 t = tmp1[(((size_t)bn * C_ + c) * H_ + h) * 12 + kw];
    sr += t.x * cs + t.y * sn;       // t * e^{-i a}
    si += t.y * cs - t.x * sn;
  }
  sr *= (1.0f / 32.0f); si *= (1.0f / 32.0f);
  const size_t zr = ((size_t)bn * 144 + kh * 12 + kw) * C2;
  z[zr + c] = sr;        z[zr + 128 + c] = si;
  zh[zr + c] = (_Float16)sr; zh[zr + 128 + c] = (_Float16)si;
}

// -------- mix GEMM (WMMA 256x256, weights async-staged) -> delta spectrum
__global__ void k_mix(const _Float16* __restrict__ zh, const _Float16* __restrict__ mw,
                      const float* __restrict__ mb, const float* __restrict__ z,
                      float* __restrict__ delta) {
  __shared__ __align__(16) _Float16 ldsw[16 * 256];
  const int nt = blockIdx.x & 15;
  const int mtg = blockIdx.x >> 4;
  stage_weights_async(ldsw, mw + (size_t)nt * 16 * 256, 16 * 256);
  const int wid = threadIdx.x >> 5, lane = threadIdx.x & 31;
  const int mt = mtg * 4 + wid;
  F8 a;
#pragma unroll
  for (int i = 0; i < 8; ++i) a.f[i] = 0.f;
  a.v = wmma_kloop_ldsB(zh, 256, ldsw, 256, mt * 16, 256, a.v);
  const int off = (lane & 16) ? 8 : 0;
  const int col = nt * 16 + (lane & 15);
#pragma unroll
  for (int i = 0; i < 8; ++i) {
    size_t idx = (size_t)(mt * 16 + i + off) * C2 + col;
    delta[idx] = a.f[i] + mb[col] - z[idx];     // Delta = mixed - original modes
  }
}

// --------------------------------------------- inverse DFT along H of Delta
__global__ void k_idfth(const float* __restrict__ delta, float2* __restrict__ tmp2) {
  const int idx = blockIdx.x * 256 + threadIdx.x;   // ((bn*128+c)*32+h)*12+kw
  const int kw = idx % 12;
  const int rest = idx / 12;
  const int h = rest & 31, c = (rest >> 5) & 127, bn = rest >> 12;
  float sr = 0.f, si = 0.f;
#pragma unroll
  for (int kh = 0; kh < 12; ++kh) {
    float cs, sn;
    __sincosf(PI2 * (float)((kh * h) & 31) / 32.0f, &sn, &cs);
    size_t zr = ((size_t)bn * 144 + kh * 12 + kw) * C2;
    float dr = delta[zr + c], di = delta[zr + 128 + c];
    sr += dr * cs - di * sn;          // d * e^{+i a}
    si += dr * sn + di * cs;
  }
  tmp2[idx] = make_float2(sr, si);
}

// --------------- out = 2*y3 + irfft_w(Delta rows), transposed to (B,C,L,H,W)
__global__ void k_final(const float* __restrict__ y3, const float2* __restrict__ tmp2,
                        float* __restrict__ out) {
  const int idx = blockIdx.x * 256 + threadIdx.x;   // ((((b*C+c)*L+l)*H+h)*W+w)
  const int w = idx & 31, h = (idx >> 5) & 31, l = (idx >> 10) & 31;
  const int c = (idx >> 15) & 127, b = idx >> 22;
  const int bn = b * L_ + l;
  const float2* th = tmp2 + (((size_t)bn * C_ + c) * H_ + h) * 12;
  float inv = th[0].x;
#pragma unroll
  for (int kw = 1; kw < 12; ++kw) {
    float cs, sn;
    __sincosf(PI2 * (float)((kw * w) & 31) / 32.0f, &sn, &cs);
    float2 t = th[kw];
    inv += 2.0f * (t.x * cs - t.y * sn);
  }
  inv *= (1.0f / 32.0f);
  float yv = y3[((size_t)bn * NTOK + h * 32 + w) * C_ + c];
  out[idx] = 2.0f * yv + inv;
}

// ---------------------------------------------------------------- launcher
extern "C" void kernel_launch(void* const* d_in, const int* in_sizes, int n_in,
                              void* d_out, int out_size, void* d_ws, size_t ws_size,
                              hipStream_t stream) {
  const float* x      = (const float*)d_in[0];
  const float* dt     = (const float*)d_in[1];
  const float* nu     = (const float*)d_in[2];
  const float* theta  = (const float*)d_in[3];
  const float* ln_g   = (const float*)d_in[4];
  const float* ln_b   = (const float*)d_in[5];
  const float* w_qkv  = (const float*)d_in[6];
  const float* w_proj = (const float*)d_in[7];
  const float* b_proj = (const float*)d_in[8];
  const float* gn_g   = (const float*)d_in[9];
  const float* gn_b   = (const float*)d_in[10];
  const float* w_g    = (const float*)d_in[11];
  const float* w_l    = (const float*)d_in[12];
  const float* w_psi  = (const float*)d_in[13];
  const float* b_psi  = (const float*)d_in[14];
  const float* gate_g = (const float*)d_in[15];
  const float* gate_b = (const float*)d_in[16];
  const float* mix_w  = (const float*)d_in[17];
  const float* mix_b  = (const float*)d_in[18];

  char* ws = (char*)d_ws;
  const size_t OFF_STATES = 0;                       // 35,651,584 B
  const size_t OFF_XT     = 35651584;                // 33,554,432 B
  const size_t OFF_QKV    = OFF_XT + 33554432;       // 50,331,648 B
  const size_t OFF_XT2    = OFF_QKV + 50331648;      // 33,554,432 B
  const size_t OFF_POOL   = OFF_XT2 + 33554432;      //  4,194,304 B
  const size_t OFF_GG     = OFF_POOL + 4194304;      //  8,388,608 B
  const size_t OFF_Z      = OFF_GG + 8388608;        //  9,437,184 B
  const size_t OFF_ZH     = OFF_Z + 9437184;         //  4,718,592 B
  const size_t OFF_DELTA  = OFF_ZH + 4718592;        //  9,437,184 B
  const size_t OFF_WH     = OFF_DELTA + 9437184;     //    327,680 B
  const size_t OFF_STATS  = OFF_WH + 327680;         //      2,048 B

  float2*    states  = (float2*)(ws + OFF_STATES);
  float*     xt      = (float*)(ws + OFF_XT);
  _Float16*  xn_h    = (_Float16*)(ws + OFF_STATES);           // alias (states dead)
  _Float16*  q_h     = (_Float16*)(ws + OFF_QKV);
  _Float16*  k_h     = (_Float16*)(ws + OFF_QKV + 16777216);
  _Float16*  v_h     = (_Float16*)(ws + OFF_QKV + 33554432);
  _Float16*  ao_h    = (_Float16*)(ws + OFF_STATES);           // alias (xn dead)
  float*     xt2     = (float*)(ws + OFF_XT2);
  float*     y2n     = (float*)(ws + OFF_QKV);                 // alias (qkv dead)
  _Float16*  y2n_h   = (_Float16*)(ws + OFF_QKV + 33554432);
  _Float16*  pooled  = (_Float16*)(ws + OFF_POOL);
  float*     gg      = (float*)(ws + OFF_GG);
  float*     y2g     = (float*)(ws + OFF_STATES);              // alias (ao dead)
  float*     y3      = (float*)(ws + OFF_XT);                  // alias (xt dead)
  float2*    tmp1    = (float2*)(ws + OFF_XT2);                // alias (xt2 dead)
  float*     z       = (float*)(ws + OFF_Z);
  _Float16*  z_h     = (_Float16*)(ws + OFF_ZH);
  float*     delta   = (float*)(ws + OFF_DELTA);
  float2*    tmp2    = (float2*)(ws + OFF_STATES);             // alias (y2g dead)
  _Float16*  wh      = (_Float16*)(ws + OFF_WH);
  _Float16*  wqkv_h  = wh;
  _Float16*  wproj_h = wh + 49152;
  _Float16*  wg_h    = wh + 65536;
  _Float16*  wl_h    = wh + 81920;
  _Float16*  mixw_h  = wh + 98304;
  float2*    stats   = (float2*)(ws + OFF_STATS);
  float*     out     = (float*)d_out;

  k_cvt    <<<640,   256, 0, stream>>>(w_qkv, w_proj, w_g, w_l, mix_w, wh);
  k_scan   <<<8192,  32,  0, stream>>>(x, dt, nu, theta, states);
  k_irfft  <<<32768, 256, 0, stream>>>(states, xt);
  k_ln     <<<8192,  256, 0, stream>>>(xt, ln_g, ln_b, xn_h);
  k_qkv    <<<24576, 128, 0, stream>>>(xn_h, wqkv_h, q_h, k_h, v_h);
  k_attn   <<<32768, 32,  0, stream>>>(q_h, k_h, v_h, ao_h);
  k_proj   <<<8192,  128, 0, stream>>>(ao_h, wproj_h, b_proj, xt, xt2);
  k_gnstats<<<256,   256, 0, stream>>>(xt2, stats);
  k_gnapply<<<32768, 256, 0, stream>>>(xt2, stats, gn_g, gn_b, y2n, y2n_h);
  k_pool   <<<8192,  256, 0, stream>>>(y2n, pooled);
  k_gg     <<<2048,  128, 0, stream>>>(pooled, wg_h, gg);
  k_gate   <<<1024,  128, 0, stream>>>(y2n_h, wl_h, gg, w_psi, b_psi, y2n, y2g);
  k_gnstats<<<256,   256, 0, stream>>>(y2g, stats);
  k_gnapply<<<32768, 256, 0, stream>>>(y2g, stats, gate_g, gate_b, y3, (_Float16*)nullptr);
  k_dftw   <<<12288, 256, 0, stream>>>(y3, tmp1);
  k_dfth   <<<4608,  256, 0, stream>>>(tmp1, z, z_h);
  k_mix    <<<2304,  128, 0, stream>>>(z_h, mixw_h, mix_b, z, delta);
  k_idfth  <<<12288, 256, 0, stream>>>(delta, tmp2);
  k_final  <<<32768, 256, 0, stream>>>(y3, tmp2, out);

  (void)in_sizes; (void)n_in; (void)out_size; (void)ws_size;
}
